// DeepSeekAtten_88605175316747
// MI455X (gfx1250) — matrix operand order repaired and checked by reading, not compiled, not run on hardware
//
#include <hip/hip_runtime.h>
#include <hip/hip_bf16.h>

typedef __attribute__((ext_vector_type(16))) _Float16 v16h;
typedef __attribute__((ext_vector_type(8)))  _Float16 v8h;
typedef __attribute__((ext_vector_type(4)))  _Float16 v4h;
typedef __attribute__((ext_vector_type(8)))  float    v8f;

static __device__ __forceinline__ v8f v8f_zero() {
  v8f z = {0.f,0.f,0.f,0.f,0.f,0.f,0.f,0.f};
  return z;
}

// ---- WMMA fragment loaders (wave32 layouts per CDNA5 ISA 7.12.2) ----

// A fragment: 16x32 f16. Source row-major [16][ld] halves, p at (row0,k0).
// lanes 0-15: M=lane, K {0..7,16..23}; lanes 16-31: M=lane-16, K {8..15,24..31}.
static __device__ __forceinline__ v16h load_a_frag(const _Float16* p, int ld) {
  int lane = threadIdx.x & 31;
  int row  = lane & 15;
  int kh   = (lane >> 4) << 3;            // 0 or 8
  const _Float16* q = p + row * ld + kh;
  v16h a;
#pragma unroll
  for (int i = 0; i < 8; ++i) a[i] = q[i];
#pragma unroll
  for (int i = 0; i < 8; ++i) a[8 + i] = q[16 + i];
  return a;
}

// B fragment from N-major storage: memory rows are N, contiguous in K.
// logical B[k][n] = p[n*ld + k]; p at (k0,n0). Per-lane: K {0..7,16..23} / {8..15,24..31}.
static __device__ __forceinline__ v16h load_bT_frag(const _Float16* p, int ld) {
  int lane = threadIdx.x & 31;
  int col  = lane & 15;
  int k0   = (lane >> 4) << 3;            // 0 or 8
  const _Float16* q = p + col * ld + k0;
  v16h b;
#pragma unroll
  for (int i = 0; i < 8; ++i) b[i] = q[i];
#pragma unroll
  for (int i = 0; i < 8; ++i) b[8 + i] = q[16 + i];
  return b;
}

// ---------------- generic tiled WMMA GEMM (f32 in, f16 compute, f32 out) -----------
// C[M,N] = A[M,K] * B[K,N], row-major, M%128==0, N%128==0, K%64==0.
#define BM 128
#define BN 128
#define BK 64
#define LDT 72   // padded K-stride in halves (144B rows, 16B aligned)

__global__ __launch_bounds__(256) void gemm_wmma_f16(
    const float* __restrict__ A, const float* __restrict__ B, float* __restrict__ C,
    int M, int N, int K, int lda, int ldb, int ldc,
    long long strideA, long long strideB, long long strideC)
{
  A += (long long)blockIdx.z * strideA;
  B += (long long)blockIdx.z * strideB;
  C += (long long)blockIdx.z * strideC;

  __shared__ _Float16 As[BM][LDT];   // [m][k]
  __shared__ _Float16 Bs[BN][LDT];   // [n][k]  (transposed tile)

  const int tid  = threadIdx.x;
  const int wave = tid >> 5;
  const int wr   = wave >> 2;        // 0..1: 64-row group
  const int wc   = wave & 3;         // 0..3: 32-col group
  const int bm   = blockIdx.y * BM;
  const int bn   = blockIdx.x * BN;

  v8f acc[4][2];
#pragma unroll
  for (int i = 0; i < 4; ++i)
#pragma unroll
    for (int j = 0; j < 2; ++j) acc[i][j] = v8f_zero();

  // A fill: thread owns row (tid>>1), k-range (tid&1)*32..+32  (8 float4 loads)
  const int arow = tid >> 1;
  const int acol = (tid & 1) << 5;
  // B fill: thread owns k-range (tid>>5)*8..+8, n-range (tid&31)*4..+4
  const int bk0  = (tid >> 5) << 3;
  const int bn0  = (tid & 31) << 2;

  for (int k0 = 0; k0 < K; k0 += BK) {
    // ---- stage A tile ----
    {
      const float4* ag = (const float4*)(A + (size_t)(bm + arow) * lda + k0 + acol);
      _Float16* dst = &As[arow][acol];
#pragma unroll
      for (int v = 0; v < 8; ++v) {
        float4 t = ag[v];
        dst[v*4+0] = (_Float16)t.x; dst[v*4+1] = (_Float16)t.y;
        dst[v*4+2] = (_Float16)t.z; dst[v*4+3] = (_Float16)t.w;
      }
    }
    // ---- stage B tile transposed: registers hold 8k x 4n block ----
    {
      _Float16 tv[8][4];
#pragma unroll
      for (int dk = 0; dk < 8; ++dk) {
        float4 t = *(const float4*)(B + (size_t)(k0 + bk0 + dk) * ldb + bn + bn0);
        tv[dk][0] = (_Float16)t.x; tv[dk][1] = (_Float16)t.y;
        tv[dk][2] = (_Float16)t.z; tv[dk][3] = (_Float16)t.w;
      }
#pragma unroll
      for (int dn = 0; dn < 4; ++dn) {
        v8h p = { tv[0][dn], tv[1][dn], tv[2][dn], tv[3][dn],
                  tv[4][dn], tv[5][dn], tv[6][dn], tv[7][dn] };
        *(v8h*)&Bs[bn0 + dn][bk0] = p;   // one packed 16B LDS store
      }
    }
    if (k0 + BK < K) {   // prefetch next tiles (global_prefetch_b8)
      __builtin_prefetch(A + (size_t)(bm + arow) * lda + (k0 + BK) + acol, 0, 1);
      __builtin_prefetch(B + (size_t)(k0 + BK + bk0) * ldb + bn + bn0, 0, 1);
    }
    __syncthreads();

#pragma unroll
    for (int kk = 0; kk < 2; ++kk) {
      const int kb = kk << 5;
      v16h af[4];
#pragma unroll
      for (int mt = 0; mt < 4; ++mt)
        af[mt] = load_a_frag(&As[wr * 64 + mt * 16][kb], LDT);
#pragma unroll
      for (int nt = 0; nt < 2; ++nt) {
        v16h bf = load_bT_frag(&Bs[wc * 32 + nt * 16][kb], LDT);
#pragma unroll
        for (int mt = 0; mt < 4; ++mt)
          acc[mt][nt] = __builtin_amdgcn_wmma_f32_16x16x32_f16(
              false, af[mt], false, bf, (short)0, acc[mt][nt], false, false);
      }
    }
    __syncthreads();
  }

  // ---- epilogue: one base pointer, int strength-reduced offsets ----
  const int lane  = tid & 31;
  const int ccol  = lane & 15;
  const int crow0 = (lane >> 4) << 3;
  float* cb = C + (size_t)(bm + wr * 64 + crow0) * ldc + (bn + wc * 32 + ccol);
#pragma unroll
  for (int mt = 0; mt < 4; ++mt)
#pragma unroll
    for (int nt = 0; nt < 2; ++nt) {
      float* cp = cb + mt * 16 * ldc + nt * 16;
      int off = 0;
#pragma unroll
      for (int r = 0; r < 8; ++r) { cp[off] = acc[mt][nt][r]; off += ldc; }
    }
}

// ---------------- copy rope part of q into q_full ----------------
__global__ void pe_copy(const float* __restrict__ q, float* __restrict__ q_full) {
  int idx  = blockIdx.x * blockDim.x + threadIdx.x;    // < 512*16*64
  int m    = idx >> 10;
  int rest = idx & 1023;
  int h    = rest >> 6;
  int j    = rest & 63;
  q_full[(size_t)m * 9216 + h * 576 + 512 + j] = q[(size_t)m * 3072 + h * 192 + 128 + j];
}

// ---------------- attention: one block per query row m ----------------
#define NDA 576     // D_ATT
#define NKV 512     // D_KV_C
#define NTK 512     // TOPK
#define NH  16
#define CHUNK 64
#define QLD 584
#define KVLD 584    // pass-1 layout [key][d]
#define KVTLD 72    // pass-2 layout [col][key], 64+8 pad
#define ATLD 520

__global__ __launch_bounds__(256) void attn_kernel(
    const float* __restrict__ q_full, const float* __restrict__ kv_cache,
    const int* __restrict__ indices, float* __restrict__ o)
{
  extern __shared__ char smem[];
  float*    sc_s  = (float*)smem;                 // [16][512] partial scores (K 0..287)
  float*    sc2_s = sc_s + NH * NTK;              // [16][512] partial scores (K 288..575)
  float*    red_s = sc2_s + NH * NTK;             // [16][16]
  _Float16* q_s   = (_Float16*)(red_s + 256);     // [16][QLD]
  _Float16* kvb   = q_s + NH * QLD;               // max(64*KVLD, 512*KVTLD) halves
  _Float16* at_s  = kvb + CHUNK * KVLD;           // [16][ATLD]

  _Float16* kv_s  = kvb;   // pass-1 view: [64][KVLD]
  _Float16* kvt_s = kvb;   // pass-2 view: [512][KVTLD]

  const int m    = blockIdx.x;
  const int tid  = threadIdx.x;
  const int wave = tid >> 5;
  const float sm_scale = 0.041666666666666664f;   // 1/sqrt(576)

  // stage q_full[m] (16 heads x 576) as f16
  {
    const float* qr = q_full + (size_t)m * (NH * NDA);
    for (int i = tid; i < NH * NDA; i += 256) {
      int h = i / NDA, d = i - h * NDA;
      q_s[h * QLD + d] = (_Float16)qr[i];
    }
  }

  // ---- pass 1: scores = q_full . kv^T (8 waves: 4 key tiles x 2 K-halves) ----
  const int gj = tid >> 2;            // key-in-chunk 0..63
  const int gp = (tid & 3) * 144;     // 144 floats per quarter row
  const int sj0 = (wave & 3) << 4;    // key tile within chunk
  const int skh = (wave >> 2) * 288;  // K half
  float* scb = (wave >> 2) ? sc2_s : sc_s;

  for (int c = 0; c < NTK / CHUNK; ++c) {
    __syncthreads();
    {
      int key = indices[(size_t)m * NTK + c * CHUNK + gj];
      const float4* src = (const float4*)(kv_cache + (size_t)key * NDA + gp);
      _Float16* dst = kv_s + gj * KVLD + gp;
#pragma unroll
      for (int v = 0; v < 36; ++v) {
        float4 t = src[v];
        dst[v*4+0] = (_Float16)t.x; dst[v*4+1] = (_Float16)t.y;
        dst[v*4+2] = (_Float16)t.z; dst[v*4+3] = (_Float16)t.w;
      }
    }
    __syncthreads();
    {
      v8f acc = v8f_zero();
      for (int k = 0; k < 288; k += 32) {
        v16h a = load_a_frag(q_s + skh + k, QLD);
        v16h b = load_bT_frag(kv_s + sj0 * KVLD + skh + k, KVLD);
        acc = __builtin_amdgcn_wmma_f32_16x16x32_f16(
            false, a, false, b, (short)0, acc, false, false);
      }
      int lane = tid & 31, col = lane & 15, r0 = (lane >> 4) << 3;
#pragma unroll
      for (int r = 0; r < 8; ++r)
        scb[(r0 + r) * NTK + c * CHUNK + sj0 + col] = acc[r];
    }
  }
  __syncthreads();

  // ---- softmax over 512 keys per head row: 16 threads per row, 32 cols each ----
  {
    int r = tid >> 4, sub = tid & 15;
    int j0 = sub * 32;
    float v[32];
    float mx = -3.0e38f;
#pragma unroll
    for (int j = 0; j < 32; ++j) {
      v[j] = (sc_s[r * NTK + j0 + j] + sc2_s[r * NTK + j0 + j]) * sm_scale;
      mx = fmaxf(mx, v[j]);
    }
    red_s[r * 16 + sub] = mx;
    __syncthreads();
    float rmx = -3.0e38f;
    for (int i = 0; i < 16; ++i) rmx = fmaxf(rmx, red_s[r * 16 + i]);
    __syncthreads();
    float s = 0.f;
#pragma unroll
    for (int j = 0; j < 32; ++j) { v[j] = __expf(v[j] - rmx); s += v[j]; }
    red_s[r * 16 + sub] = s;
    __syncthreads();
    float rs = 0.f;
    for (int i = 0; i < 16; ++i) rs += red_s[r * 16 + i];
    float inv = 1.f / rs;
#pragma unroll
    for (int j = 0; j < 32; ++j)
      at_s[r * ATLD + j0 + j] = (_Float16)(v[j] * inv);
  }

  // ---- pass 2: O[16,512] = attn[16,512] x kv_c[512,512]; kv staged transposed ----
  v8f oacc[4];
#pragma unroll
  for (int i = 0; i < 4; ++i) oacc[i] = v8f_zero();
  const int n0  = wave * 64;
  const int tj0 = (tid & 15) << 2;    // 4 keys per thread
  const int td0 = (tid >> 4) << 5;    // 32 cols per thread

  for (int c = 0; c < NTK / CHUNK; ++c) {
    __syncthreads();
    {
      int key[4];
#pragma unroll
      for (int j = 0; j < 4; ++j)
        key[j] = indices[(size_t)m * NTK + c * CHUNK + tj0 + j];
#pragma unroll
      for (int db = 0; db < 4; ++db) {          // 8 cols per sub-block
        _Float16 vals[4][8];
#pragma unroll
        for (int j = 0; j < 4; ++j) {
          const float4* src =
              (const float4*)(kv_cache + (size_t)key[j] * NDA + td0 + db * 8);
          float4 a = src[0], b = src[1];
          vals[j][0]=(_Float16)a.x; vals[j][1]=(_Float16)a.y;
          vals[j][2]=(_Float16)a.z; vals[j][3]=(_Float16)a.w;
          vals[j][4]=(_Float16)b.x; vals[j][5]=(_Float16)b.y;
          vals[j][6]=(_Float16)b.z; vals[j][7]=(_Float16)b.w;
        }
#pragma unroll
        for (int d = 0; d < 8; ++d) {
          v4h p = { vals[0][d], vals[1][d], vals[2][d], vals[3][d] };
          *(v4h*)&kvt_s[(td0 + db * 8 + d) * KVTLD + tj0] = p;  // packed 8B store
        }
      }
    }
    __syncthreads();
#pragma unroll
    for (int kk = 0; kk < 2; ++kk) {
      v16h a = load_a_frag(at_s + c * CHUNK + kk * 32, ATLD);
#pragma unroll
      for (int nt = 0; nt < 4; ++nt) {
        v16h b = load_bT_frag(kvt_s + (n0 + nt * 16) * KVTLD + kk * 32, KVTLD);
        oacc[nt] = __builtin_amdgcn_wmma_f32_16x16x32_f16(
            false, a, false, b, (short)0, oacc[nt], false, false);
      }
    }
  }

  // store o[m][h][col]
  int lane = tid & 31, col = lane & 15, r0 = (lane >> 4) << 3;
  float* ob = o + (size_t)m * (NH * NKV) + r0 * NKV + n0 + col;
#pragma unroll
  for (int nt = 0; nt < 4; ++nt) {
    int off = nt * 16;
#pragma unroll
    for (int r = 0; r < 8; ++r) { ob[off] = oacc[nt][r]; off += NKV; }
  }
}

extern "C" void kernel_launch(void* const* d_in, const int* in_sizes, int n_in,
                              void* d_out, int out_size, void* d_ws, size_t ws_size,
                              hipStream_t stream) {
  (void)in_sizes; (void)n_in; (void)out_size; (void)ws_size;
  const float* x        = (const float*)d_in[0];  // [512,7168]
  const float* W_cqkv   = (const float*)d_in[1];  // [7168,2112]
  const float* W_uq     = (const float*)d_in[2];  // [1536,3072]
  const float* W_qk     = (const float*)d_in[3];  // [16,128,512]
  const float* kv_cache = (const float*)d_in[4];  // [4096,576]
  const float* W_o1     = (const float*)d_in[5];  // [16,512,128]
  const float* W_oproj  = (const float*)d_in[6];  // [2048,7168]
  const int*   indices  = (const int*)d_in[7];    // [512,512]
  float* out = (float*)d_out;                     // [512,7168]

  float* q_c    = (float*)d_ws;                       // 512*1536
  float* q      = q_c    + (size_t)512 * 1536;        // 512*3072
  float* q_full = q      + (size_t)512 * 3072;        // 512*9216
  float* o      = q_full + (size_t)512 * 9216;        // 512*8192
  float* o2     = o      + (size_t)512 * 8192;        // 512*2048

  dim3 blk(256);

  // 1) q_c = x @ W_cqkv[:, 512:2048]   (kv_c_new / rope slices of c are dead)
  gemm_wmma_f16<<<dim3(1536/128, 512/128, 1), blk, 0, stream>>>(
      x, W_cqkv + 512, q_c, 512, 1536, 7168, 7168, 2112, 1536, 0, 0, 0);

  // 2) q = q_c @ W_uq
  gemm_wmma_f16<<<dim3(3072/128, 4, 1), blk, 0, stream>>>(
      q_c, W_uq, q, 512, 3072, 1536, 1536, 3072, 3072, 0, 0, 0);

  // 3) q_full[:,:512] per head: q_nope[h] @ W_qk[h]
  gemm_wmma_f16<<<dim3(512/128, 4, 16), blk, 0, stream>>>(
      q, W_qk, q_full, 512, 512, 128, 3072, 512, 9216,
      192LL, 128LL * 512, 576LL);

  // 4) q_full[:,512:576] = q_pe
  pe_copy<<<dim3((512 * 16 * 64) / 256), blk, 0, stream>>>(q, q_full);

  // 5) attention (one block per m)
  size_t shmem = (size_t)(2 * NH * NTK + 256) * sizeof(float)
               + (size_t)(NH * QLD + CHUNK * KVLD + NH * ATLD) * sizeof(_Float16);
  attn_kernel<<<dim3(512), blk, shmem, stream>>>(q_full, kv_cache, indices, o);

  // 6) o2 per head: o[h] @ W_o1[h]
  gemm_wmma_f16<<<dim3(1, 4, 16), blk, 0, stream>>>(
      o, W_o1, o2, 512, 128, 512, 8192, 128, 2048,
      512LL, 512LL * 128, 128LL);

  // 7) out = o2 @ W_oproj
  gemm_wmma_f16<<<dim3(7168/128, 4, 1), blk, 0, stream>>>(
      o2, W_oproj, out, 512, 7168, 2048, 2048, 7168, 7168, 0, 0, 0);
}
